// CapsNet_37976100831588
// MI455X (gfx1250) — matrix operand order repaired and checked by reading, not compiled, not run on hardware
//
#include <hip/hip_runtime.h>
#include <hip/hip_bf16.h>
#include <math.h>

typedef __bf16 bf16_t;
typedef __attribute__((ext_vector_type(16))) __bf16 v16bf;
typedef __attribute__((ext_vector_type(8)))  __bf16 v8bf;
typedef __attribute__((ext_vector_type(8)))  float  v8f;
typedef unsigned int v4u __attribute__((ext_vector_type(4)));
typedef int v8i __attribute__((ext_vector_type(8)));
typedef int v4i __attribute__((ext_vector_type(4)));

#define HAS_TDM (__has_builtin(__builtin_amdgcn_tensor_load_to_lds) && \
                 __has_builtin(__builtin_amdgcn_s_wait_tensorcnt))

__device__ __forceinline__ bf16_t tobf(float f) { return (bf16_t)f; }

__device__ __forceinline__ v16bf make_v16(v8bf lo, v8bf hi) {
    v16bf r;
#pragma unroll
    for (int i = 0; i < 8; i++) { r[i] = lo[i]; r[8 + i] = hi[i]; }
    return r;
}

// ---------------- weight repacks (fp32 -> bf16, GEMM-friendly layouts) ----------------

__global__ void repack_w1(const float* __restrict__ w1, bf16_t* __restrict__ w1bf) {
    int i = blockIdx.x * blockDim.x + threadIdx.x;          // [256][96], zero-pad k>=81
    if (i >= 256 * 96) return;
    int oc = i / 96, k = i - oc * 96;
    w1bf[i] = tobf(k < 81 ? w1[oc * 81 + k] : 0.0f);
}

__global__ void repack_w2(const float* __restrict__ w2, bf16_t* __restrict__ w2bf) {
    int i = blockIdx.x * blockDim.x + threadIdx.x;          // natural OIHW == Bt[n][k]
    if (i >= 256 * 20736) return;
    w2bf[i] = tobf(w2[i]);
}

__global__ void repack_W(const float* __restrict__ W, bf16_t* __restrict__ Wt) {
    // W[32][288][160] -> Wt[32][160][288]
    int i = blockIdx.x * blockDim.x + threadIdx.x;
    if (i >= 32 * 288 * 160) return;
    int n = i / (288 * 160); int r = i - n * (288 * 160);
    int d = r / 160; int kk = r - d * 160;
    Wt[((size_t)n * 160 + kk) * 288 + d] = tobf(W[i]);
}

// ---------------- conv1: implicit GEMM, M=400 pos, N=256 oc, K=81 (pad 96) ----------------

__global__ __launch_bounds__(512, 1) void conv1_wmma(const float* __restrict__ data,
                                                     const bf16_t* __restrict__ w1bf,
                                                     const float* __restrict__ b1,
                                                     bf16_t* __restrict__ h1bf) {
    __shared__ bf16_t lds_a[16 * 104];                      // [16][96], stride 104 (pad)
    const int b = blockIdx.x;
    const int t = threadIdx.x;
    const int wave = t >> 5;                                // 16 waves -> oc tiles
    const int lane = t & 31;
    const int half = lane >> 4;
    const int lcol = lane & 15;
    const int oc = wave * 16 + lcol;

    v16bf bfrag[3];
#pragma unroll
    for (int s = 0; s < 3; s++)
        bfrag[s] = *(const v16bf*)(w1bf + oc * 96 + s * 32 + half * 16);
    const float bias = b1[oc];
    const float* img = data + (size_t)b * 784;

    for (int mt = 0; mt < 25; mt++) {
        __syncthreads();
#pragma unroll
        for (int i = 0; i < 3; i++) {                       // stage 16x96 im2col tile
            int e = t + i * 512;
            int m = e / 96, k = e - m * 96;
            int p = mt * 16 + m;
            int oy = p / 20, ox = p - oy * 20;
            float v = 0.0f;
            if (k < 81) {
                int ky = k / 9, kx = k - ky * 9;
                v = img[(oy + ky) * 28 + (ox + kx)];
            }
            lds_a[m * 104 + k] = tobf(v);
        }
        __syncthreads();
        v8f c = {};
#pragma unroll
        for (int s = 0; s < 3; s++) {
            const bf16_t* base = lds_a + lcol * 104 + s * 32 + half * 16;
            v16bf a = make_v16(*(const v8bf*)base, *(const v8bf*)(base + 8));
            c = __builtin_amdgcn_wmma_f32_16x16x32_bf16(false, a, false, bfrag[s],
                                                        (short)0, c, false, false);
        }
        int p0 = mt * 16 + half * 8;                        // 8 contiguous positions
        bf16_t out8[8];
#pragma unroll
        for (int r = 0; r < 8; r++) out8[r] = tobf(c[r] + bias);
        *(v8bf*)(h1bf + ((size_t)b * 256 + oc) * 400 + p0) = *(v8bf*)out8;
    }
}

// ---------------- conv2: implicit GEMM, M=36(->48) pos, N=256 oc, K=20736 ----------------
// Whole per-image activation slab (256x400 bf16 = 204.8KB) is DMA'd into LDS once by the
// Tensor Data Mover, then the im2col A-tile is double-buffered (1 barrier / k-step).
// Output written directly in capsule layout u_pre[b][32][288] (fused reshape/transpose).

__global__ __launch_bounds__(512, 1) void conv2_wmma(const bf16_t* __restrict__ h1bf,
                                                     const bf16_t* __restrict__ w2bf,
                                                     const float* __restrict__ b2,
                                                     float* __restrict__ u_pre) {
    extern __shared__ char smem_raw[];
    bf16_t* lds_img = (bf16_t*)smem_raw;                    // 102400 elems (204800 B)
    bf16_t* lds_a0  = (bf16_t*)(smem_raw + 204800);         // [48][40] bf16
    bf16_t* lds_a1  = lds_a0 + 48 * 40;

    const int b = blockIdx.x;
    const int t = threadIdx.x;
    const int wave = t >> 5;
    const int lane = t & 31;
    const int half = lane >> 4;
    const int lcol = lane & 15;
    const int oc = wave * 16 + lcol;
    const bf16_t* img = h1bf + (size_t)b * 102400;
    const bf16_t* wrow = w2bf + (size_t)oc * 20736;

    // ---- bring the whole image into LDS ----
#if HAS_TDM
    if (wave == 0) {                                        // wave-uniform branch
        unsigned long long ga = (unsigned long long)(uintptr_t)img;
        unsigned int      la = (unsigned int)(uintptr_t)lds_img;
        v4u g0; v8i g1; v4i g2; v4i g3; v8i g4;
        g0[0] = 1u;                                         // count=1
        g0[1] = la;                                         // lds_addr
        g0[2] = (unsigned int)ga;                           // global_addr[31:0]
        g0[3] = ((unsigned int)(ga >> 32) & 0x01FFFFFFu) | 0x80000000u; // addr[56:32]|type=2
        g1[0] = (int)(1u << 16);                            // data_size=1 (2B), mask=0
        g1[1] = (int)(400u << 16);                          // tensor_dim0 = 400 (lo16)
        g1[2] = (int)(256u << 16);                          // tensor_dim1 = 256 (lo16)
        g1[3] = (int)(400u << 16);                          // tile_dim0 = 400
        g1[4] = (int)256u;                                  // tile_dim1 = 256, tile_dim2=0
        g1[5] = (int)400u;                                  // tensor_dim0_stride = 400
        g1[6] = (int)(0x9000u << 16);                       // tensor_dim1_stride lo16 (102400)
        g1[7] = (int)1u;                                    // tensor_dim1_stride hi
        g2[0] = 0; g2[1] = 0; g2[2] = 0; g2[3] = 0;
        g3[0] = 0; g3[1] = 0; g3[2] = 0; g3[3] = 0;
        g4[0] = 0; g4[1] = 0; g4[2] = 0; g4[3] = 0;
        g4[4] = 0; g4[5] = 0; g4[6] = 0; g4[7] = 0;
        __builtin_amdgcn_tensor_load_to_lds(g0, g1, g2, g3, g4, 0);
        __builtin_amdgcn_s_wait_tensorcnt(0);
    }
#else
    {                                                       // cooperative fallback copy
        const unsigned int* src = (const unsigned int*)img;
        unsigned int* dst = (unsigned int*)lds_img;
        for (int i = t; i < 25600; i += 512) dst[i] = src[i];
    }
#endif
    __syncthreads();

    // ---- per-thread im2col staging state (3 elements of the 48x32 tile each) ----
    int ci[3], r[3], st[3], baseP[3]; bool valid[3];
#pragma unroll
    for (int i = 0; i < 3; i++) {
        int e = t + i * 512;
        int m = e >> 5, kk = e & 31;                        // m wave-uniform, kk per-lane
        st[i] = m * 40 + kk;
        valid[i] = (m < 36);
        int mm = valid[i] ? m : 0;
        int oy = mm / 6, ox = mm - oy * 6;
        baseP[i] = oy * 40 + ox * 2;                        // (oy*2)*20 + ox*2
        ci[i] = 0; r[i] = kk;                               // k = kk at ks=0
    }

    // prologue: stage tile ks=0 into buf0
#pragma unroll
    for (int i = 0; i < 3; i++) {
        int ky = r[i] / 9, kx = r[i] - ky * 9;
        bf16_t v = valid[i] ? lds_img[ci[i] * 400 + baseP[i] + ky * 20 + kx] : tobf(0.0f);
        lds_a0[st[i]] = v;
    }
    __syncthreads();

    v8f c0 = {}, c1 = {}, c2 = {};
    for (int ks = 0; ks < 648; ks++) {
        const int p = ks & 1;
        bf16_t* bufp = p ? lds_a1 : lds_a0;
        bf16_t* bufn = p ? lds_a0 : lds_a1;
        const bool more = (ks + 1 < 648);

        bf16_t nxt[3];
        if (more) {                                         // gather tile ks+1 (DS loads)
#pragma unroll
            for (int i = 0; i < 3; i++) {
                r[i] += 32;
                if (r[i] >= 81) { r[i] -= 81; ci[i]++; }
                int ky = r[i] / 9, kx = r[i] - ky * 9;
                nxt[i] = valid[i] ? lds_img[ci[i] * 400 + baseP[i] + ky * 20 + kx]
                                  : tobf(0.0f);
            }
        }

        v16bf bfv = *(const v16bf*)(wrow + ks * 32 + half * 16);
#pragma unroll
        for (int mt = 0; mt < 3; mt++) {
            const bf16_t* base = bufp + (mt * 16 + lcol) * 40 + half * 16;
            v16bf a = make_v16(*(const v8bf*)base, *(const v8bf*)(base + 8));
            v8f& c = (mt == 0) ? c0 : ((mt == 1) ? c1 : c2);
            c = __builtin_amdgcn_wmma_f32_16x16x32_bf16(false, a, false, bfv,
                                                        (short)0, c, false, false);
        }

        if (more) {
#pragma unroll
            for (int i = 0; i < 3; i++) bufn[st[i]] = nxt[i];
        }
        __syncthreads();                                    // one barrier per k-step
    }

    const float bias = b2[oc];
    const int ncap = oc >> 3, c8 = oc & 7;
    float* urow = u_pre + ((size_t)b * 32 + ncap) * 288;
#pragma unroll
    for (int mt = 0; mt < 3; mt++) {
        v8f c = (mt == 0) ? c0 : ((mt == 1) ? c1 : c2);
#pragma unroll
        for (int rr = 0; rr < 8; rr++) {
            int m = mt * 16 + half * 8 + rr;
            if (m < 36) urow[m * 8 + c8] = c[rr] + bias;    // d = pos*8 + (oc&7)
        }
    }
}

// ---------------- squash: per-(b,capsule) norm over 288, emit bf16 for routing GEMM ----------------

__global__ void squash_kernel(const float* __restrict__ u_pre, bf16_t* __restrict__ u_bf) {
    int gw = (blockIdx.x * blockDim.x + threadIdx.x) >> 5;  // one wave per (b,n)
    int lane = threadIdx.x & 31;
    if (gw >= 1024 * 32) return;
    const float* row = u_pre + (size_t)gw * 288;
    float s = 0.0f;
#pragma unroll
    for (int i = 0; i < 9; i++) { float v = row[lane + i * 32]; s += v * v; }
#pragma unroll
    for (int off = 16; off > 0; off >>= 1) s += __shfl_xor(s, off, 32);
    float scale = s / (1.0f + s) / sqrtf(s);
    bf16_t* orow = u_bf + (size_t)gw * 288;
#pragma unroll
    for (int i = 0; i < 9; i++) orow[lane + i * 32] = tobf(row[lane + i * 32] * scale);
}

// ---------------- u_hat = u @ W per capsule: 32 GEMMs, M=1024, N=160, K=288 ----------------

__global__ __launch_bounds__(32) void uhat_wmma(const bf16_t* __restrict__ u_bf,
                                                const bf16_t* __restrict__ Wt,
                                                float* __restrict__ u_hat) {
    const int mt = blockIdx.x;                              // 64 batch tiles
    const int jt = blockIdx.y;                              // 10 output tiles
    const int n  = blockIdx.z;                              // 32 capsules
    const int lane = threadIdx.x;
    const int half = lane >> 4;
    const int lcol = lane & 15;
    const bf16_t* arow = u_bf + ((size_t)(mt * 16 + lcol) * 32 + n) * 288;
    const bf16_t* brow = Wt + ((size_t)n * 160 + jt * 16 + lcol) * 288;
    v8f c = {};
#pragma unroll
    for (int s = 0; s < 9; s++) {
        const bf16_t* ab = arow + s * 32 + half * 16;
        v16bf a  = make_v16(*(const v8bf*)ab, *(const v8bf*)(ab + 8));
        v16bf bv = *(const v16bf*)(brow + s * 32 + half * 16);
        c = __builtin_amdgcn_wmma_f32_16x16x32_bf16(false, a, false, bv,
                                                    (short)0, c, false, false);
    }
    int col = jt * 16 + lcol;
#pragma unroll
    for (int r = 0; r < 8; r++) {
        int m = mt * 16 + half * 8 + r;
        u_hat[((size_t)m * 32 + n) * 160 + col] = c[r];
    }
}

// ---------------- routing (3 iterations, deterministic reductions) ----------------

__global__ void zero_small(float* __restrict__ logits) {
    int t = threadIdx.x;
    if (t < 320) logits[t] = 0.0f;
}

__global__ __launch_bounds__(320) void route_iter(const float* __restrict__ u_hat,
                                                  const float* __restrict__ logits,
                                                  float* __restrict__ s_buf,
                                                  float* __restrict__ out,
                                                  int final_iter) {
    __shared__ float c_sh[320];
    __shared__ float v_sh[160];
    const int b = blockIdx.x;
    const int t = threadIdx.x;
    if (t < 10) {                                           // softmax over n per column j
        float mx = -1e30f;
        for (int n = 0; n < 32; n++) mx = fmaxf(mx, logits[n * 10 + t]);
        float e[32]; float sum = 0.0f;
        for (int n = 0; n < 32; n++) { e[n] = expf(logits[n * 10 + t] - mx); sum += e[n]; }
        float inv = 1.0f / sum;
        for (int n = 0; n < 32; n++) c_sh[n * 10 + t] = e[n] * inv;
    }
    __syncthreads();
    if (t < 160) {                                          // v[b][j][k]
        int j = t >> 4, k = t & 15;
        float v = 0.0f;
        for (int n = 0; n < 32; n++)
            v += u_hat[((size_t)b * 32 + n) * 160 + j * 16 + k] * c_sh[n * 10 + j];
        v_sh[t] = v;
        if (final_iter) out[10240 + b * 160 + t] = v;
    }
    __syncthreads();
    if (final_iter) {
        if (t < 10) {
            float s = 0.0f;
            for (int k = 0; k < 16; k++) { float v = v_sh[t * 16 + k]; s += v * v; }
            out[b * 10 + t] = sqrtf(s);
        }
    } else {                                                // per-(b,n,j) agreement
        int n = t / 10, j = t - n * 10;
        float s = 0.0f;
        for (int k = 0; k < 16; k++) {
            float x = u_hat[((size_t)b * 32 + n) * 160 + j * 16 + k] * v_sh[j * 16 + k];
            s += x * x;
        }
        s_buf[(size_t)b * 320 + t] = sqrtf(s);
    }
}

__global__ __launch_bounds__(320) void logits_update(float* __restrict__ logits,
                                                     const float* __restrict__ s_buf) {
    int t = threadIdx.x;
    float s = 0.0f;
    for (int b = 0; b < 1024; b++) s += s_buf[(size_t)b * 320 + t];
    logits[t] += s * (1.0f / 1024.0f);
}

// ---------------- host launch ----------------

extern "C" void kernel_launch(void* const* d_in, const int* in_sizes, int n_in,
                              void* d_out, int out_size, void* d_ws, size_t ws_size,
                              hipStream_t stream) {
    const float* data = (const float*)d_in[0];
    const float* w1   = (const float*)d_in[1];
    const float* b1   = (const float*)d_in[2];
    const float* w2   = (const float*)d_in[3];
    const float* b2   = (const float*)d_in[4];
    const float* W    = (const float*)d_in[5];
    float* out = (float*)d_out;

    char* ws = (char*)d_ws;
    size_t off = 0;
    auto alloc = [&](size_t bytes) -> void* {
        void* p = ws + off;
        off = (off + bytes + 255) & ~(size_t)255;
        return p;
    };
    bf16_t* h1bf   = (bf16_t*)alloc((size_t)1024 * 256 * 400 * 2); // 209.7 MB
    bf16_t* w1bf   = (bf16_t*)alloc((size_t)256 * 96 * 2);
    bf16_t* w2bf   = (bf16_t*)alloc((size_t)256 * 20736 * 2);      // 10.6 MB
    float*  u_pre  = (float*) alloc((size_t)1024 * 9216 * 4);      // 37.7 MB
    bf16_t* u_bf   = (bf16_t*)alloc((size_t)1024 * 9216 * 2);      // 18.9 MB
    bf16_t* Wt     = (bf16_t*)alloc((size_t)32 * 160 * 288 * 2);   // 2.9 MB
    float*  u_hat  = (float*) alloc((size_t)1024 * 32 * 160 * 4);  // 21.0 MB
    float*  logits = (float*) alloc(320 * 4);
    float*  s_buf  = (float*) alloc((size_t)1024 * 320 * 4);       // 1.3 MB

    repack_w1<<<(256 * 96 + 255) / 256, 256, 0, stream>>>(w1, w1bf);
    repack_w2<<<(256 * 20736 + 255) / 256, 256, 0, stream>>>(w2, w2bf);
    repack_W<<<(32 * 288 * 160 + 255) / 256, 256, 0, stream>>>(W, Wt);

    conv1_wmma<<<1024, 512, 0, stream>>>(data, w1bf, b1, h1bf);
    const size_t conv2_smem = 204800 + 2 * 48 * 40 * 2;            // 212480 B < 320 KB
    conv2_wmma<<<1024, 512, conv2_smem, stream>>>(h1bf, w2bf, b2, u_pre);
    squash_kernel<<<4096, 256, 0, stream>>>(u_pre, u_bf);
    uhat_wmma<<<dim3(64, 10, 32), 32, 0, stream>>>(u_bf, Wt, u_hat);

    zero_small<<<1, 320, 0, stream>>>(logits);
    route_iter<<<1024, 320, 0, stream>>>(u_hat, logits, s_buf, out, 0);
    logits_update<<<1, 320, 0, stream>>>(logits, s_buf);
    route_iter<<<1024, 320, 0, stream>>>(u_hat, logits, s_buf, out, 0);
    logits_update<<<1, 320, 0, stream>>>(logits, s_buf);
    route_iter<<<1024, 320, 0, stream>>>(u_hat, logits, s_buf, out, 1);
}
